// RoutingCapsule_18476949307591
// MI455X (gfx1250) — matrix operand order repaired and checked by reading, not compiled, not run on hardware
//
#include <hip/hip_runtime.h>
#include <hip/hip_bf16.h>

typedef __attribute__((ext_vector_type(16))) __bf16 v16bf;
typedef __attribute__((ext_vector_type(8)))  float  v8f;
typedef __attribute__((address_space(3))) __bf16 lds_bf16_t;

#define B_SZ  128
#define P_SZ  2048
#define NPC   32
#define NC    16
#define CL    32
#define KDIM  512   // NC*CL
#define PSPLIT 8    // P-range splits for routing occupancy

// ---------------------------------------------------------------------------
// Kernel 1: squash primary capsules, write bf16 t.
// ---------------------------------------------------------------------------
__global__ __launch_bounds__(256) void squash_primary(
    const float* __restrict__ x, __bf16* __restrict__ t) {
  int row = blockIdx.x * blockDim.x + threadIdx.x;   // 0 .. B*P-1
  if (row >= B_SZ * P_SZ) return;

  const float4* xr = (const float4*)(x + (size_t)row * NPC);
  float4 v[NPC / 4];
  float ss = 0.0f;
#pragma unroll
  for (int j = 0; j < NPC / 4; ++j) {
    v[j] = xr[j];
    ss += v[j].x * v[j].x + v[j].y * v[j].y + v[j].z * v[j].z + v[j].w * v[j].w;
  }
  float scale = ss / (1.0f + ss) * rsqrtf(ss);

  __bf16* tr = t + (size_t)row * NPC;
#pragma unroll
  for (int j = 0; j < NPC / 4; ++j) {
    tr[4 * j + 0] = (__bf16)(v[j].x * scale);
    tr[4 * j + 1] = (__bf16)(v[j].y * scale);
    tr[4 * j + 2] = (__bf16)(v[j].z * scale);
    tr[4 * j + 3] = (__bf16)(v[j].w * scale);
  }
}

// ---------------------------------------------------------------------------
// Kernel 2: per-position GEMM  u[b, p, k] = sum_n t[b,p,n] * w[p,n,k]
// One workgroup (8 waves) per p.
//   A = weight tile: M=16 output features x K=32 npc
//   B = t tile     : K=32 npc x N=16 batch columns
//   D : lane owns one batch row, 8 CONSECUTIVE k -> one b128 store per tile.
// ---------------------------------------------------------------------------
__global__ __launch_bounds__(256) void capsule_gemm(
    const __bf16* __restrict__ t, const float* __restrict__ w,
    __bf16* __restrict__ u) {
  __shared__ __align__(16) __bf16 lds_w[KDIM][NPC];  // [k_out][npc] 32 KB (A src)
  __shared__ __align__(16) __bf16 lds_t[B_SZ][NPC];  // [batch][npc]  8 KB (B src)

  const int p   = blockIdx.x;
  const int tid = threadIdx.x;

  // Stage weight[p] (32 x 512 fp32) -> LDS transposed [k_out][npc] bf16.
  // Process two npc-rows (k, k+1) at a time: float4 global loads (coalesced in
  // n), pack bf16 pairs (v_cvt_pk_bf16_f32), b32 LDS stores.
  {
    const float4* wp4 = (const float4*)(w + (size_t)p * NPC * KDIM);
    for (int i = tid; i < (NPC / 2) * (KDIM / 4); i += 256) {  // 2048
      int k2 = i / (KDIM / 4);       // 0..15 -> k = 2*k2
      int n4 = i % (KDIM / 4);       // 0..127 -> n = 4*n4
      float4 a = wp4[(2 * k2 + 0) * (KDIM / 4) + n4];
      float4 b = wp4[(2 * k2 + 1) * (KDIM / 4) + n4];
      union { unsigned int w32; __bf16 h[2]; } pk;
      pk.h[0] = (__bf16)a.x; pk.h[1] = (__bf16)b.x;
      *(unsigned int*)&lds_w[n4 * 4 + 0][2 * k2] = pk.w32;
      pk.h[0] = (__bf16)a.y; pk.h[1] = (__bf16)b.y;
      *(unsigned int*)&lds_w[n4 * 4 + 1][2 * k2] = pk.w32;
      pk.h[0] = (__bf16)a.z; pk.h[1] = (__bf16)b.z;
      *(unsigned int*)&lds_w[n4 * 4 + 2][2 * k2] = pk.w32;
      pk.h[0] = (__bf16)a.w; pk.h[1] = (__bf16)b.w;
      *(unsigned int*)&lds_w[n4 * 4 + 3][2 * k2] = pk.w32;
    }
  }
  // Stage t[:, p, :] (128 x 32 bf16)
  for (int i = tid; i < B_SZ * NPC; i += 256) {
    int b = i / NPC;
    int k = i % NPC;
    lds_t[b][k] = t[((size_t)b * P_SZ + p) * NPC + k];
  }
  __syncthreads();

  const int wave = tid / 32;        // N-tile: batch rows [wave*16, wave*16+16)
  const int lane = tid % 32;
  const int half = lane / 16;
  const int l16  = lane % 16;

  // B fragment: lane owns batch column b = wave*16+l16; half h holds npc 8h..8h+7 and 16+8h..16+8h+7
  union { v16bf v; uint4 q[2]; } bfrag;
  const uint4* trow = (const uint4*)&lds_t[wave * 16 + l16][0];
  bfrag.q[0] = trow[half];
  bfrag.q[1] = trow[2 + half];

  const int bb = wave * 16 + l16;   // this lane's batch row (fixed)

  for (int mt = 0; mt < KDIM / 16; ++mt) {
    // A fragment (16-bit A 16x32 interleave): row m = mt*16 + l16
    union { v16bf v; uint4 q[2]; } afrag;
    const uint4* wrow = (const uint4*)&lds_w[mt * 16 + l16][0];
    afrag.q[0] = wrow[half];
    afrag.q[1] = wrow[2 + half];

    v8f c = {};
    c = __builtin_amdgcn_wmma_f32_16x16x32_bf16(
        false, afrag.v, false, bfrag.v, (short)0, c, false, false);

    // D: VGPR r -> k = mt*16 + half*8 + r (consecutive), col N=l16 -> batch bb
    union { uint4 q; __bf16 h[8]; } st;
#pragma unroll
    for (int r = 0; r < 8; ++r) st.h[r] = (__bf16)c[r];
    const int kbase = mt * 16 + half * 8;
    *(uint4*)&u[((size_t)bb * P_SZ + p) * KDIM + kbase] = st.q;
  }
}

// ---------------------------------------------------------------------------
// Kernel 3: zero the s accumulator (B x KDIM fp32).
// ---------------------------------------------------------------------------
__global__ __launch_bounds__(256) void zero_s(float* __restrict__ s) {
  s[blockIdx.x * 256 + threadIdx.x] = 0.0f;
}

// ---------------------------------------------------------------------------
// Kernel 4: one routing pass over a P-slice, reading u ONCE (async to LDS).
//   iter 0: c = 1/16 (softmax of zero bias)
//   iter i>0: logits[p,n] = dot(u[p,n,:], vsum[n,:]) with vsum = v0 (+ v1)
//   s_partial[b,k] += sum_p c[p,n(k)] * u[p,k]   (global fp32 atomics)
// Grid: (B_SZ, PSPLIT), 256 threads.
// ---------------------------------------------------------------------------
__global__ __launch_bounds__(256) void routing_pass(
    const __bf16* __restrict__ u, const float* __restrict__ v0buf,
    const float* __restrict__ v1buf, float* __restrict__ s_buf, int iter) {
  __shared__ __align__(16) __bf16 lds_u[16][KDIM];  // 16 KB u chunk
  __shared__ float lds_bv[16][NC];
  __shared__ float lds_c[16][NC];
  __shared__ float lds_vs[KDIM];

  const int b   = blockIdx.x;
  const int tid = threadIdx.x;
  const int kk  = tid * 2;          // two adjacent k in same capsule n
  const int n   = tid / 16;         // = kk / CL
  const int pp  = tid / NC;         // chunk-local p (0..15)
  const int nn  = tid % NC;

  const __bf16* ub = u + (size_t)b * P_SZ * KDIM;

  // LDS byte offsets for this thread's 4 async b128 destinations
  lds_bf16_t* lbase = (lds_bf16_t*)&lds_u[0][0];
  unsigned ldsoff = (unsigned)(unsigned long long)(lbase) + (unsigned)tid * 16u;

  if (iter == 1) {
    float2 a = *(const float2*)&v0buf[(size_t)b * KDIM + kk];
    lds_vs[kk] = a.x; lds_vs[kk + 1] = a.y;
  } else if (iter == 2) {
    float2 a = *(const float2*)&v0buf[(size_t)b * KDIM + kk];
    float2 c = *(const float2*)&v1buf[(size_t)b * KDIM + kk];
    lds_vs[kk] = a.x + c.x; lds_vs[kk + 1] = a.y + c.y;
  }
  __syncthreads();

  float acc0 = 0.0f, acc1 = 0.0f;
  const int pbeg = blockIdx.y * (P_SZ / PSPLIT);
  const int pend = pbeg + (P_SZ / PSPLIT);

  for (int p0 = pbeg; p0 < pend; p0 += 16) {
    // Stage u[b, p0..p0+15, :] (16 KB) -> LDS with async b128 copies.
    // 4 x (256 threads x 16B) covers 1024 uint4.
    {
      const __bf16* g0 = ub + (size_t)p0 * KDIM + (size_t)tid * 8;
#pragma unroll
      for (int j = 0; j < 4; ++j) {
        asm volatile("global_load_async_to_lds_b128 %0, %1, off"
                     :: "v"(ldsoff + j * 4096u), "v"(g0 + j * 2048)
                     : "memory");
      }
      asm volatile("s_wait_asynccnt 0x0" ::: "memory");
    }
    __syncthreads();

    // prefetch next chunk while we compute on this one
    if (p0 + 16 < pend) __builtin_prefetch(ub + (size_t)(p0 + 16) * KDIM, 0, 3);

    if (iter == 0) {
      lds_c[pp][nn] = 1.0f / 16.0f;
    } else {
      // logits via dot(u-chunk row, vsum) on bf16 pairs
      float dot = 0.0f;
      const unsigned int* ur = (const unsigned int*)&lds_u[pp][nn * CL];
      const float* vv = &lds_vs[nn * CL];
#pragma unroll
      for (int c2 = 0; c2 < CL / 2; ++c2) {
        union { unsigned int w; __bf16 h[2]; } pr; pr.w = ur[c2];
        dot += (float)pr.h[0] * vv[2 * c2] + (float)pr.h[1] * vv[2 * c2 + 1];
      }
      lds_bv[pp][nn] = dot;
      __syncthreads();
      float m = lds_bv[pp][0];
#pragma unroll
      for (int j = 1; j < NC; ++j) m = fmaxf(m, lds_bv[pp][j]);
      float sum = 0.0f;
#pragma unroll
      for (int j = 0; j < NC; ++j) sum += __expf(lds_bv[pp][j] - m);
      lds_c[pp][nn] = __expf(lds_bv[pp][nn] - m) / sum;
    }
    __syncthreads();

    // weighted accumulation over the 16 p's of this chunk
#pragma unroll 4
    for (int q = 0; q < 16; ++q) {
      union { unsigned int w; __bf16 h[2]; } pr;
      pr.w = *(const unsigned int*)&lds_u[q][kk];
      float cq = lds_c[q][n];
      acc0 += cq * (float)pr.h[0];
      acc1 += cq * (float)pr.h[1];
    }
    __syncthreads();
  }

  atomicAdd(&s_buf[(size_t)b * KDIM + kk],     acc0);
  atomicAdd(&s_buf[(size_t)b * KDIM + kk + 1], acc1);
}

// ---------------------------------------------------------------------------
// Kernel 5: v = squash(s) per (b, n). One block per b.
// ---------------------------------------------------------------------------
__global__ __launch_bounds__(256) void squash_v(
    const float* __restrict__ s_buf, float* __restrict__ vout) {
  __shared__ float lds_s[KDIM];
  __shared__ float lds_ss[NC];
  const int b = blockIdx.x, tid = threadIdx.x, kk = tid * 2, n = tid / 16;

  float2 sv = *(const float2*)&s_buf[(size_t)b * KDIM + kk];
  lds_s[kk] = sv.x; lds_s[kk + 1] = sv.y;
  __syncthreads();
  if (tid < NC) {
    float ss = 0.0f;
#pragma unroll
    for (int c2 = 0; c2 < CL; ++c2) {
      float v = lds_s[tid * CL + c2];
      ss += v * v;
    }
    lds_ss[tid] = ss / (1.0f + ss) * rsqrtf(ss);
  }
  __syncthreads();
  float sc = lds_ss[n];
  float2 out; out.x = sv.x * sc; out.y = sv.y * sc;
  *(float2*)&vout[(size_t)b * KDIM + kk] = out;
}

// ---------------------------------------------------------------------------
extern "C" void kernel_launch(void* const* d_in, const int* in_sizes, int n_in,
                              void* d_out, int out_size, void* d_ws, size_t ws_size,
                              hipStream_t stream) {
  const float* x = (const float*)d_in[0];   // (128, 8, 16, 16, 32) fp32
  const float* w = (const float*)d_in[1];   // (2048, 32, 512) fp32
  float* out = (float*)d_out;               // (128, 16, 32) fp32

  char* ws = (char*)d_ws;
  const size_t t_bytes = (size_t)B_SZ * P_SZ * NPC * sizeof(__bf16);    //  16 MB
  const size_t u_bytes = (size_t)B_SZ * P_SZ * KDIM * sizeof(__bf16);   // 268 MB
  const size_t v_bytes = (size_t)B_SZ * KDIM * sizeof(float);           // 256 KB
  __bf16* t    = (__bf16*)ws;
  __bf16* u    = (__bf16*)(ws + t_bytes);
  float*  v0   = (float*)(ws + t_bytes + u_bytes);
  float*  v1   = (float*)(ws + t_bytes + u_bytes + v_bytes);
  float*  sbuf = (float*)(ws + t_bytes + u_bytes + 2 * v_bytes);

  squash_primary<<<(B_SZ * P_SZ) / 256, 256, 0, stream>>>(x, t);
  capsule_gemm<<<P_SZ, 256, 0, stream>>>(t, w, u);

  for (int iter = 0; iter < 3; ++iter) {
    zero_s<<<(B_SZ * KDIM) / 256, 256, 0, stream>>>(sbuf);
    routing_pass<<<dim3(B_SZ, PSPLIT), 256, 0, stream>>>(u, v0, v1, sbuf, iter);
    float* dst = (iter == 0) ? v0 : (iter == 1) ? v1 : out;
    squash_v<<<B_SZ, 256, 0, stream>>>(sbuf, dst);
  }
}
